// BertSelfAttention_438086664286
// MI455X (gfx1250) — matrix operand order repaired and checked by reading, not compiled, not run on hardware
//
#include <hip/hip_runtime.h>
#include <hip/hip_bf16.h>
#include <math.h>

// ---------------------------------------------------------------------------
// BERT self-attention for MI455X (gfx1250, wave32, WMMA + TDM).
//
// Roofline: the probs output is 512 MB f32 -> >=22us at 23.3 TB/s HBM; math is
// only ~86 GFLOP (trivial for f16 WMMA). All operands are L2-resident (192 MB),
// so after the HBM floor the next constraint is L2 traffic:
//   - QKV GEMM: 128x128 tiles  -> ~1.5 GB L2 re-reads (vs 3 GB at 64x64)
//   - attention: 32-query stripes halve K/V re-reads (~1 GB) and make the
//     P*V stage map 1:1 onto the 8 waves (no cross-wave reduction)
// probs are written exactly once, coalesced; Q/K/V are f16 in workspace.
//
// Workspace (48 MB):
//   Q  : [8192][1024] f16 (row = b*S+s, col = h*64+d)
//   K  : [8192][1024] f16
//   Vt : [8][16][64][1024] f16 (transposed so the P*V B-fragment is contiguous)
// ---------------------------------------------------------------------------

#define BB 8
#define SS 1024
#define DD 1024
#define HH 16
#define HD 64
#define MTOT (BB * SS) // 8192

typedef __attribute__((ext_vector_type(16))) _Float16     v16h;
typedef __attribute__((ext_vector_type(8)))  _Float16     v8h;
typedef __attribute__((ext_vector_type(8)))  float        v8f;
typedef __attribute__((ext_vector_type(4)))  float        v4f;
typedef __attribute__((ext_vector_type(4)))  unsigned int u32x4;
typedef __attribute__((ext_vector_type(8)))  int          i32x8;
typedef __attribute__((ext_vector_type(4)))  int          i32x4;

#if defined(__has_builtin)
#if __has_builtin(__builtin_amdgcn_tensor_load_to_lds)
#define HAVE_TDM 1
#endif
#endif
#ifndef HAVE_TDM
#define HAVE_TDM 0
#endif

// ============================================================================
// Kernel 1: Y = X*W + b, 128x128 tile per 256-thread block (8 waves in 4x2).
// Each wave owns a 32x64 subtile: 2 A-frags x 4 B-frags -> 8 accumulators.
// The W tile is DMA'd raw (f32) into LDS by the Tensor Data Mover, then
// cooperatively converted/transposed to the f16 B-fragment layout.
// z==0 -> Q, z==1 -> K (both [M,D] f16), z==2 -> V transposed [b,h,d,s].
// ============================================================================
__global__ __launch_bounds__(256)
void qkv_proj_kernel(const float* __restrict__ X,
                     const float* __restrict__ Wq, const float* __restrict__ bq,
                     const float* __restrict__ Wk, const float* __restrict__ bk,
                     const float* __restrict__ Wv, const float* __restrict__ bv,
                     _Float16* __restrict__ Q, _Float16* __restrict__ K,
                     _Float16* __restrict__ Vt)
{
    __shared__ __align__(16) _Float16 As[128 * 32];   // [row][k]  8 KB
    __shared__ __align__(16) _Float16 BsT[128 * 32];  // [n][k]    8 KB
#if HAVE_TDM
    __shared__ __align__(16) float Wraw[32 * 128];    // raw TDM tile, 16 KB
#endif

    const int z = blockIdx.z;
    const float* __restrict__ W    = (z == 0) ? Wq : (z == 1) ? Wk : Wv;
    const float* __restrict__ bias = (z == 0) ? bq : (z == 1) ? bk : bv;

    const int mbase = blockIdx.y * 128;
    const int nbase = blockIdx.x * 128;
    const int t    = threadIdx.x;
    const int lane = t & 31;
    const int w    = t >> 5;

    const int m0    = (w & 3) * 32;   // wave's 32 rows within tile
    const int ncol0 = (w >> 2) * 64;  // wave's 64 cols within tile

    v8f acc[2][4] = {};

    for (int k0 = 0; k0 < DD; k0 += 32) {
        __syncthreads();
        // ---- stage A tile: 128 rows x 32 k, f32 -> f16 (2 threads/row) ----
        {
            const int row = t >> 1;
            const int kc  = (t & 1) * 16;
            const float* src = X + (size_t)(mbase + row) * DD + k0 + kc;
            const v4f x0 = *(const v4f*)(src + 0);
            const v4f x1 = *(const v4f*)(src + 4);
            const v4f x2 = *(const v4f*)(src + 8);
            const v4f x3 = *(const v4f*)(src + 12);
            v8h h0, h1;
#pragma unroll
            for (int j = 0; j < 4; ++j) {
                h0[j] = (_Float16)x0[j]; h0[4 + j] = (_Float16)x1[j];
                h1[j] = (_Float16)x2[j]; h1[4 + j] = (_Float16)x3[j];
            }
            *(v8h*)&As[row * 32 + kc]     = h0;
            *(v8h*)&As[row * 32 + kc + 8] = h1;
        }
#if HAVE_TDM
        // ---- stage B tile via Tensor Data Mover: 32 x 128 f32, raw ----
        if (w == 0) {
            const unsigned long long ga =
                (unsigned long long)(const void*)(W + (size_t)k0 * DD + nbase);
            const unsigned int ldsa = (unsigned int)(size_t)(void*)Wraw;
            // D# group0: count=1 valid | lds_addr | global_addr[56:0] | type=2
            u32x4 g0 = { 1u, ldsa,
                         (unsigned int)(ga & 0xffffffffu),
                         (unsigned int)((ga >> 32) & 0x01ffffffu) | (2u << 30) };
            // D# group1: data_size=4B; tensor 1024x1024; tile 128x32; stride 1024
            i32x8 g1 = { 0x00020000,        // wg_mask=0 | data_size=2 (4B)
                         0x04000000,        // tensor_dim0(=1024) low16 << 16
                         0x04000000,        // tensor_dim1(=1024) low16 << 16
                         (128 << 16),       // tile_dim0 = 128
                         32,                // tile_dim1 = 32
                         1024,              // tensor_dim0_stride low32
                         0, 0 };
            i32x4 gz = { 0, 0, 0, 0 };
#if __clang_major__ >= 23
            i32x8 gz8 = { 0, 0, 0, 0, 0, 0, 0, 0 };
            __builtin_amdgcn_tensor_load_to_lds(g0, g1, gz, gz, gz8, 0);
#else
            __builtin_amdgcn_tensor_load_to_lds(g0, g1, gz, gz, 0);
#endif
            asm volatile("s_wait_tensorcnt 0x0" ::: "memory");
        }
        __syncthreads();
        // ---- convert raw f32 tile -> f16 transposed B layout ----
        {
            const int kk = t >> 3;           // 0..31
            const int nc = (t & 7) * 16;     // 0,16,...,112
#pragma unroll
            for (int j = 0; j < 16; ++j)
                BsT[(nc + j) * 32 + kk] = (_Float16)Wraw[kk * 128 + nc + j];
        }
        __syncthreads();
#else
        // ---- fallback: cooperative staging of W tile, transposed ----
        {
            const int kk = t >> 3;
            const int nc = (t & 7) * 16;
            const float* src = W + (size_t)(k0 + kk) * DD + nbase + nc;
            __builtin_prefetch(src + 32 * DD, 0, 0);
#pragma unroll
            for (int j = 0; j < 16; ++j)
                BsT[(nc + j) * 32 + kk] = (_Float16)src[j];
        }
        __syncthreads();
#endif
        // ---- A fragments (16x32 f16) ----
        v16h a[2];
#pragma unroll
        for (int rb = 0; rb < 2; ++rb) {
            const int row = m0 + rb * 16 + (lane & 15);
            const int kb  = (lane < 16) ? 0 : 16;
            const v8h c0 = *(const v8h*)&As[row * 32 + kb];
            const v8h c1 = *(const v8h*)&As[row * 32 + kb + 8];
#pragma unroll
            for (int j = 0; j < 8; ++j) { a[rb][j] = c0[j]; a[rb][8 + j] = c1[j]; }
        }
        // ---- B fragments (32x16 f16) ----
        const int kb2 = (lane < 16) ? 0 : 16;
        v16h bf[4];
#pragma unroll
        for (int ct = 0; ct < 4; ++ct)
            bf[ct] = *(const v16h*)&BsT[(ncol0 + ct * 16 + (lane & 15)) * 32 + kb2];

#pragma unroll
        for (int rb = 0; rb < 2; ++rb)
#pragma unroll
            for (int ct = 0; ct < 4; ++ct)
                acc[rb][ct] = __builtin_amdgcn_wmma_f32_16x16x32_f16(
                    false, a[rb], false, bf[ct], (short)0, acc[rb][ct], false, false);
    }

    // ---- epilogue: bias add + f16 store ----
    const int mlo = (lane < 16) ? 0 : 8;
#pragma unroll
    for (int ct = 0; ct < 4; ++ct) {
        const int col = nbase + ncol0 + ct * 16 + (lane & 15);
        const float bs = bias[col];
#pragma unroll
        for (int rb = 0; rb < 2; ++rb) {
#pragma unroll
            for (int r = 0; r < 8; ++r) {
                const int m = mbase + m0 + rb * 16 + mlo + r;
                const float y = acc[rb][ct][r] + bs;
                if (z < 2) {
                    _Float16* O = (z == 0) ? Q : K;
                    O[(size_t)m * DD + col] = (_Float16)y;
                } else {
                    const int bidx = m >> 10, s = m & 1023;
                    Vt[(((size_t)bidx * HH + (col >> 6)) * HD + (col & 63)) * SS + s] =
                        (_Float16)y;
                }
            }
        }
    }
}

// ============================================================================
// Kernel 2: fused attention for one (b,h) x 32-query stripe per block.
//   Phase 1: scores(32x1024) = Q K^T / 8  (WMMA) -> LDS (128 KB f32)
//   Phase 2: softmax rows in LDS (+mask, +head_mask), coalesced probs store
//   Phase 3: ctx(32x64) = P * V; 8 output tiles map 1:1 onto the 8 waves
// ============================================================================
__global__ __launch_bounds__(256)
void attention_kernel(const _Float16* __restrict__ Q,
                      const _Float16* __restrict__ Kx,
                      const _Float16* __restrict__ Vt,
                      const unsigned char* __restrict__ mask,
                      const float* __restrict__ head_mask,
                      float* __restrict__ ctx,
                      float* __restrict__ probs)
{
    __shared__ __align__(16) float sc[32 * 1024]; // 128 KB score/prob stripe

    const int bh = blockIdx.y;       // b*H + h
    const int b  = bh >> 4;
    const int h  = bh & 15;
    const int q0 = blockIdx.x * 32;  // query stripe base
    const int t = threadIdx.x, lane = t & 31, w = t >> 5;
    const int hbase = h * HD;
    const float hm = head_mask[h];

    // ---------------- Phase 1: scores = Q K^T * (1/8) ----------------
    v16h aq[2][2]; // [row-block][k-step]
    {
        const int kb = (lane < 16) ? 0 : 16;
#pragma unroll
        for (int rb = 0; rb < 2; ++rb) {
            const int row = q0 + rb * 16 + (lane & 15);
            const _Float16* qrow = Q + (size_t)(b * SS + row) * DD + hbase;
#pragma unroll
            for (int ks = 0; ks < 2; ++ks) {
                const v8h c0 = *(const v8h*)(qrow + ks * 32 + kb);
                const v8h c1 = *(const v8h*)(qrow + ks * 32 + kb + 8);
#pragma unroll
                for (int j = 0; j < 8; ++j) {
                    aq[rb][ks][j] = c0[j]; aq[rb][ks][8 + j] = c1[j];
                }
            }
        }
    }
    // Each wave covers 8 key-tiles of 16 keys (8 x 8 x 16 = 1024 keys).
    const int kb2 = (lane < 16) ? 0 : 16;
#pragma unroll 1
    for (int kt = w * 8; kt < w * 8 + 8; ++kt) {
        const int key0 = kt * 16;
        const _Float16* krow = Kx + (size_t)(b * SS + key0 + (lane & 15)) * DD + hbase;
        v8f acc[2] = {};
#pragma unroll
        for (int ks = 0; ks < 2; ++ks) {
            const v16h bk = *(const v16h*)(krow + ks * 32 + kb2);
#pragma unroll
            for (int rb = 0; rb < 2; ++rb)
                acc[rb] = __builtin_amdgcn_wmma_f32_16x16x32_f16(
                    false, aq[rb][ks], false, bk, (short)0, acc[rb], false, false);
        }
        const int key = key0 + (lane & 15);
        const int mlo = (lane < 16) ? 0 : 8;
#pragma unroll
        for (int rb = 0; rb < 2; ++rb) {
#pragma unroll
            for (int r = 0; r < 8; ++r) {
                const int m = rb * 16 + mlo + r;
                float s = acc[rb][r] * 0.125f; // 1/sqrt(HD)
                if (mask[((size_t)b * SS + (q0 + m)) * SS + key]) s = -INFINITY;
                sc[m * 1024 + key] = s;
            }
        }
    }
    __syncthreads();

    // ---------------- Phase 2: softmax + probs writeback ----------------
#pragma unroll 1
    for (int rr = 0; rr < 4; ++rr) {
        const int m = w * 4 + rr;            // 8 waves x 4 rows = 32 rows
        float* row = &sc[m * 1024];
        float mx = -INFINITY;
        for (int j = lane; j < 1024; j += 32) mx = fmaxf(mx, row[j]);
#pragma unroll
        for (int o = 16; o; o >>= 1) mx = fmaxf(mx, __shfl_xor(mx, o));
        float sum = 0.f;
        for (int j = lane; j < 1024; j += 32) {
            const float e = __expf(row[j] - mx);
            row[j] = e;
            sum += e;
        }
#pragma unroll
        for (int o = 16; o; o >>= 1) sum += __shfl_xor(sum, o);
        const float inv = hm / sum; // fold head_mask into probs (ref semantics)
        float* prow = probs + ((size_t)bh * SS + (q0 + m)) * SS;
        for (int j = lane; j < 1024; j += 32) {
            const float p = row[j] * inv;
            row[j] = p;       // keep normalized probs in LDS for P*V
            prow[j] = p;      // the 512 MB store: once, coalesced
        }
    }
    __syncthreads();

    // ---------------- Phase 3: ctx = P * V ----------------
    // 2 row-blocks x 4 dim-tiles = 8 output tiles -> one per wave, full K.
    const int rb = w >> 2;  // query row-block
    const int nt = w & 3;   // dim tile (16 dims)
    v8f acc = {};
    const _Float16* vrow = Vt + ((size_t)bh * HD + nt * 16 + (lane & 15)) * SS;
    const int rowb = (rb * 16 + (lane & 15)) * 1024;
#pragma unroll 2
    for (int kk = 0; kk < 1024; kk += 32) {
        const int kb = kk + ((lane < 16) ? 0 : 16);
        const v4f x0 = *(const v4f*)&sc[rowb + kb];
        const v4f x1 = *(const v4f*)&sc[rowb + kb + 4];
        const v4f x2 = *(const v4f*)&sc[rowb + kb + 8];
        const v4f x3 = *(const v4f*)&sc[rowb + kb + 12];
        v16h a;
#pragma unroll
        for (int j = 0; j < 4; ++j) {
            a[j]      = (_Float16)x0[j];
            a[4 + j]  = (_Float16)x1[j];
            a[8 + j]  = (_Float16)x2[j];
            a[12 + j] = (_Float16)x3[j];
        }
        const v16h bv = *(const v16h*)(vrow + kk + kb2); // contiguous keys (V^T)
        acc = __builtin_amdgcn_wmma_f32_16x16x32_f16(false, a, false, bv,
                                                     (short)0, acc, false, false);
    }
    const int dim = hbase + nt * 16 + (lane & 15);
    const int mlo = (lane < 16) ? 0 : 8;
#pragma unroll
    for (int r = 0; r < 8; ++r) {
        const int qrow = q0 + rb * 16 + mlo + r;
        ctx[((size_t)(b * SS + qrow)) * DD + dim] = acc[r];
    }
}

// ============================================================================
extern "C" void kernel_launch(void* const* d_in, const int* in_sizes, int n_in,
                              void* d_out, int out_size, void* d_ws, size_t ws_size,
                              hipStream_t stream) {
    const float*         X    = (const float*)d_in[0];
    const unsigned char* mask = (const unsigned char*)d_in[1]; // bool tensor
    const float*         hmsk = (const float*)d_in[2];
    const float*         Wq   = (const float*)d_in[3];
    const float*         bq   = (const float*)d_in[4];
    const float*         Wk   = (const float*)d_in[5];
    const float*         bk   = (const float*)d_in[6];
    const float*         Wv   = (const float*)d_in[7];
    const float*         bv   = (const float*)d_in[8];

    float* out   = (float*)d_out;
    float* ctx   = out;                                  // [B,S,D]
    float* probs = out + (size_t)BB * SS * DD;           // [B,H,S,S]

    // Workspace: Q | K | V^T, f16, 16 MB each (requires ws_size >= 48 MB).
    _Float16* Qw = (_Float16*)d_ws;
    _Float16* Kw = Qw + (size_t)MTOT * DD;
    _Float16* Vw = Kw + (size_t)MTOT * DD;

    dim3 g1(DD / 128, MTOT / 128, 3); // 8 x 64 x 3 blocks
    qkv_proj_kernel<<<g1, 256, 0, stream>>>(X, Wq, bq, Wk, bk, Wv, bv, Qw, Kw, Vw);

    dim3 g2(SS / 32, BB * HH);        // 32 x 128 blocks
    attention_kernel<<<g2, 256, 0, stream>>>(Qw, Kw, Vw, mask, hmsk, ctx, probs);
}
